// MultiHeadAttention_55808805044756
// MI455X (gfx1250) — matrix operand order repaired and checked by reading, not compiled, not run on hardware
//
#include <hip/hip_runtime.h>

// ---------------------------------------------------------------------------
// MI455X (gfx1250) multi-head attention, bf16 WMMA pipeline.
//  B=4, S=2048, D=1024, H=16, hd=64
// ---------------------------------------------------------------------------

typedef __attribute__((ext_vector_type(16))) __bf16 bf16x16;
typedef __attribute__((ext_vector_type(8)))  __bf16 bf16x8;
typedef __attribute__((ext_vector_type(8)))  float  f32x8;
typedef __attribute__((ext_vector_type(4)))  float  f32x4;

#define D_MODEL 1024
#define NHEAD   16
#define HDIM    64
#define SEQ     2048
#define BATCH   4
#define NROWS   (BATCH * SEQ)   // 8192
#define TSTR    40              // LDS row stride in bf16 elems (32 + 8 pad, keeps 16B align)

static __device__ __forceinline__ bf16x16 join8(bf16x8 lo, bf16x8 hi) {
  return __builtin_shufflevector(lo, hi, 0,1,2,3,4,5,6,7,8,9,10,11,12,13,14,15);
}

static __device__ __forceinline__ f32x8 wmma_bf16(bf16x16 a, bf16x16 b, f32x8 c) {
  // 8 args: (neg_a, A, neg_b, B, c_mod, C, reuse_a, reuse_b)
  return __builtin_amdgcn_wmma_f32_16x16x32_bf16(false, a, false, b, (short)0, c, false, false);
}

static __device__ __forceinline__ float redmax16(float v) {
#pragma unroll
  for (int m = 1; m < 16; m <<= 1) v = fmaxf(v, __shfl_xor(v, m, 32));
  return v;
}
static __device__ __forceinline__ float redsum16(float v) {
#pragma unroll
  for (int m = 1; m < 16; m <<= 1) v += __shfl_xor(v, m, 32);
  return v;
}

// ---------------------------------------------------------------------------
// Y[m,n] = sum_k A[m,k] * W[n,k]   (torch Linear, y = x @ W^T)
// A: [NROWS, D_MODEL] (fp32 or bf16), W: [D_MODEL, D_MODEL] fp32 row-major,
// Y: [NROWS, D_MODEL] (bf16 or fp32).
// Block: 256 threads = 8 waves; 128x128 output tile; K tiled by 32 via LDS.
// Wave grid 4(M) x 2(N): each wave -> 32 rows x 64 cols = 2x4 WMMA tiles.
// ---------------------------------------------------------------------------
template <typename AT, typename OT>
__global__ __launch_bounds__(256)
void gemm_xwt(const AT* __restrict__ A, const float* __restrict__ W,
              OT* __restrict__ Y)
{
  __shared__ __bf16 As[128 * TSTR];
  __shared__ __bf16 Bs[128 * TSTR];

  const int t    = threadIdx.x;
  const int wave = t >> 5;
  const int lane = t & 31;
  const int l16  = lane & 15;
  const int kg   = lane >> 4;        // 0 or 1 (which 16-lane half)

  const int mblk = blockIdx.x * 128;
  const int nblk = blockIdx.y * 128;

  const int waveM = (wave & 3) * 32; // 4 waves along M
  const int waveN = (wave >> 2) * 64;// 2 waves along N

  f32x8 acc[2][4] = {};

  const int lrow = t >> 1;           // 0..127 (tile row this thread fills)
  const int lcol = (t & 1) * 16;     // 0 or 16 (16 elems per fill)

  for (int kk = 0; kk < D_MODEL; kk += 32) {
    // ---- fill A tile (convert fp32 -> bf16 if needed) ----
    {
      bf16x8 lo, hi;
      const AT* src = A + (size_t)(mblk + lrow) * D_MODEL + kk + lcol;
      if constexpr (sizeof(AT) == 4) {
        const f32x4* s4 = (const f32x4*)src;
        f32x4 a0 = s4[0], a1 = s4[1], a2 = s4[2], a3 = s4[3];
#pragma unroll
        for (int i = 0; i < 4; ++i) {
          lo[i]     = (__bf16)a0[i];
          lo[4 + i] = (__bf16)a1[i];
          hi[i]     = (__bf16)a2[i];
          hi[4 + i] = (__bf16)a3[i];
        }
      } else {
        const bf16x8* s8 = (const bf16x8*)src;
        lo = s8[0];
        hi = s8[1];
      }
      *(bf16x8*)&As[lrow * TSTR + lcol]     = lo;
      *(bf16x8*)&As[lrow * TSTR + lcol + 8] = hi;
    }
    // ---- fill W tile (rows of W == output columns), fp32 -> bf16 ----
    {
      const float* src = W + (size_t)(nblk + lrow) * D_MODEL + kk + lcol;
      const f32x4* s4 = (const f32x4*)src;
      f32x4 a0 = s4[0], a1 = s4[1], a2 = s4[2], a3 = s4[3];
      bf16x8 lo, hi;
#pragma unroll
      for (int i = 0; i < 4; ++i) {
        lo[i]     = (__bf16)a0[i];
        lo[4 + i] = (__bf16)a1[i];
        hi[i]     = (__bf16)a2[i];
        hi[4 + i] = (__bf16)a3[i];
      }
      *(bf16x8*)&Bs[lrow * TSTR + lcol]     = lo;
      *(bf16x8*)&Bs[lrow * TSTR + lcol + 8] = hi;
    }
    __syncthreads();

    // ---- A fragments: 16x32 bf16; lane<16 holds K 0..7|16..23, lane>=16 K 8..15|24..31
    bf16x16 afr[2];
#pragma unroll
    for (int mt = 0; mt < 2; ++mt) {
      const __bf16* p = &As[(waveM + mt * 16 + l16) * TSTR];
      afr[mt] = join8(*(const bf16x8*)(p + kg * 8),
                      *(const bf16x8*)(p + 16 + kg * 8));
    }
    // ---- B fragments: 32x16 bf16; lane holds col N=l16, K kg*8..+7 | 16+kg*8..+7
    bf16x16 bfr[4];
#pragma unroll
    for (int nt = 0; nt < 4; ++nt) {
      const __bf16* p = &Bs[(waveN + nt * 16 + l16) * TSTR];
      bfr[nt] = join8(*(const bf16x8*)(p + kg * 8), *(const bf16x8*)(p + 16 + kg * 8));
    }

#pragma unroll
    for (int mt = 0; mt < 2; ++mt)
#pragma unroll
      for (int nt = 0; nt < 4; ++nt)
        acc[mt][nt] = wmma_bf16(afr[mt], bfr[nt], acc[mt][nt]);

    __syncthreads();
  }

  // ---- epilogue: C layout = VGPR r, lanes0-15 M=r, lanes16-31 M=8+r ----
#pragma unroll
  for (int mt = 0; mt < 2; ++mt) {
#pragma unroll
    for (int nt = 0; nt < 4; ++nt) {
      const int rowb = mblk + waveM + mt * 16 + kg * 8;
      const int col  = nblk + waveN + nt * 16 + l16;
#pragma unroll
      for (int r = 0; r < 8; ++r) {
        float v = acc[mt][nt][r];
        OT* dst = Y + (size_t)(rowb + r) * D_MODEL + col;
        if constexpr (sizeof(OT) == 2) *dst = (OT)(__bf16)v;
        else                           *dst = (OT)v;
      }
    }
  }
}

// ---------------------------------------------------------------------------
// Flash attention over bf16 Q/K/V (workspace), fp32 softmax, bf16 ctx out.
// grid = (S/128, B*H); block = 256 (8 waves); wave owns 16 queries.
// Key loop: 32 keys/iter -> QK^T: 4 WMMAs; online softmax; PV: 4 WMMAs.
// ---------------------------------------------------------------------------
__global__ __launch_bounds__(256)
void flash_attn(const __bf16* __restrict__ Q, const __bf16* __restrict__ K,
                const __bf16* __restrict__ V, __bf16* __restrict__ CTX)
{
  __shared__ __bf16 Vt[HDIM * TSTR];       // V^T tile: [e][key], 64 x 32(+pad)
  __shared__ __bf16 Pl[8 * 16 * TSTR];     // per-wave P staging: 16 x 32(+pad)

  const int t    = threadIdx.x;
  const int wave = t >> 5;
  const int lane = t & 31;
  const int l16  = lane & 15;
  const int kg   = lane >> 4;

  const int bh = blockIdx.y;
  const int b  = bh >> 4;
  const int h  = bh & 15;
  const size_t rowbase = (size_t)b * SEQ;
  const size_t headoff = (size_t)h * HDIM;

  const int q0 = blockIdx.x * 128 + wave * 16;

  // Q A-fragments for the wave's 16 queries (d split 2 x K=32)
  bf16x16 qa[2];
#pragma unroll
  for (int c = 0; c < 2; ++c) {
    const __bf16* qp = Q + (rowbase + q0 + l16) * D_MODEL + headoff + c * 32;
    qa[c] = join8(*(const bf16x8*)(qp + kg * 8),
                  *(const bf16x8*)(qp + 16 + kg * 8));
  }

  f32x8 ctx[4] = {};
  float mrow[8], lsum[8];
#pragma unroll
  for (int r = 0; r < 8; ++r) { mrow[r] = -1e30f; lsum[r] = 0.0f; }

  for (int j = 0; j < SEQ; j += 32) {
    // ---- stage V^T tile (block-cooperative): Vt[e][key] ----
    {
      const int key = t >> 3;            // 0..31
      const int e0  = (t & 7) * 8;       // 0..56
      const __bf16* vp = V + (rowbase + j + key) * D_MODEL + headoff + e0;
      bf16x8 v = *(const bf16x8*)vp;
#pragma unroll
      for (int i = 0; i < 8; ++i) Vt[(e0 + i) * TSTR + key] = v[i];
    }
    __syncthreads();

    // ---- scores: S = Q K^T for 32 keys (2 N-tiles of 16) ----
    f32x8 s[2];
#pragma unroll
    for (int kt = 0; kt < 2; ++kt) {
      const __bf16* kp = K + (rowbase + j + kt * 16 + l16) * D_MODEL + headoff;
      bf16x16 kbA = join8(*(const bf16x8*)(kp + kg * 8),
                          *(const bf16x8*)(kp + 16 + kg * 8));
      bf16x16 kbB = join8(*(const bf16x8*)(kp + 32 + kg * 8),
                          *(const bf16x8*)(kp + 48 + kg * 8));
      f32x8 z = {};
      z = wmma_bf16(qa[0], kbA, z);
      z = wmma_bf16(qa[1], kbB, z);
      s[kt] = z;
    }

    // ---- online softmax update (fp32), rows live in 16-lane groups ----
    float corr[8];
#pragma unroll
    for (int r = 0; r < 8; ++r) {
      float v0 = s[0][r] * 0.125f;       // 1/sqrt(64)
      float v1 = s[1][r] * 0.125f;
      float mx = redmax16(fmaxf(v0, v1));
      float mnew = fmaxf(mrow[r], mx);
      float c = __expf(mrow[r] - mnew);
      float p0 = __expf(v0 - mnew);
      float p1 = __expf(v1 - mnew);
      float rs = redsum16(p0 + p1);
      lsum[r] = lsum[r] * c + rs;
      mrow[r] = mnew;
      corr[r] = c;
      s[0][r] = p0;
      s[1][r] = p1;
    }
#pragma unroll
    for (int nt = 0; nt < 4; ++nt)
#pragma unroll
      for (int r = 0; r < 8; ++r) ctx[nt][r] *= corr[r];

    // ---- C-layout -> A-layout transpose of P through per-wave LDS ----
    __bf16* pw = &Pl[wave * 16 * TSTR];
#pragma unroll
    for (int kt = 0; kt < 2; ++kt)
#pragma unroll
      for (int r = 0; r < 8; ++r)
        pw[(r + kg * 8) * TSTR + kt * 16 + l16] = (__bf16)s[kt][r];

    bf16x16 pa;
    {
      const __bf16* pr = pw + l16 * TSTR;
      pa = join8(*(const bf16x8*)(pr + kg * 8),
                 *(const bf16x8*)(pr + 16 + kg * 8));
    }

    // ---- ctx += P V : B frag cols from transposed-V LDS (table order) ----
#pragma unroll
    for (int nt = 0; nt < 4; ++nt) {
      const __bf16* vb = &Vt[(nt * 16 + l16) * TSTR];
      bf16x16 bfr = join8(*(const bf16x8*)(vb + kg * 8), *(const bf16x8*)(vb + 16 + kg * 8));
      ctx[nt] = wmma_bf16(pa, bfr, ctx[nt]);
    }
    __syncthreads();   // before next V^T staging overwrites Vt
  }

  // ---- normalize and emit ctx (bf16) in [B*S, D] with d = h*64+e ----
#pragma unroll
  for (int nt = 0; nt < 4; ++nt) {
#pragma unroll
    for (int r = 0; r < 8; ++r) {
      float inv = 1.0f / lsum[r];
      float v = ctx[nt][r] * inv;
      size_t row = rowbase + q0 + r + kg * 8;
      CTX[row * D_MODEL + headoff + nt * 16 + l16] = (__bf16)v;
    }
  }
}

// ---------------------------------------------------------------------------
extern "C" void kernel_launch(void* const* d_in, const int* in_sizes, int n_in,
                              void* d_out, int out_size, void* d_ws, size_t ws_size,
                              hipStream_t stream) {
  const float* X  = (const float*)d_in[0];
  const float* Wq = (const float*)d_in[1];
  const float* Wk = (const float*)d_in[2];
  const float* Wv = (const float*)d_in[3];
  const float* Wo = (const float*)d_in[4];
  float* out = (float*)d_out;

  const size_t mat = (size_t)NROWS * D_MODEL;   // 8M bf16 elems = 16MB each
  __bf16* Qw = (__bf16*)d_ws;
  __bf16* Kw = Qw + mat;
  __bf16* Vw = Kw + mat;
  __bf16* Cw = Vw + mat;

  dim3 blk(256);
  dim3 gGemm(NROWS / 128, D_MODEL / 128);       // 64 x 8
  dim3 gAttn(SEQ / 128, BATCH * NHEAD);         // 16 x 64

  gemm_xwt<float, __bf16><<<gGemm, blk, 0, stream>>>(X, Wq, Qw);
  gemm_xwt<float, __bf16><<<gGemm, blk, 0, stream>>>(X, Wk, Kw);
  gemm_xwt<float, __bf16><<<gGemm, blk, 0, stream>>>(X, Wv, Vw);
  flash_attn<<<gAttn, blk, 0, stream>>>(Qw, Kw, Vw, Cw);
  gemm_xwt<__bf16, float><<<gGemm, blk, 0, stream>>>(Cw, Wo, out);
}
